// Block_42580305773047
// MI455X (gfx1250) — matrix operand order repaired and checked by reading, not compiled, not run on hardware
//
#include <hip/hip_runtime.h>
#include <hip/hip_bf16.h>

// ---------------------------------------------------------------------------
// RWKV block on gfx1250: bf16 WMMA GEMMs + f32 WKV scan
// ---------------------------------------------------------------------------

typedef __bf16 bf16;
typedef __attribute__((ext_vector_type(16))) __bf16        v16bf;
typedef __attribute__((ext_vector_type(8)))  float         v8f;
typedef __attribute__((ext_vector_type(4)))  float         f32x4;
typedef __attribute__((ext_vector_type(4)))  __bf16        bf16x4;
typedef __attribute__((ext_vector_type(4)))  unsigned int  u32x4;

static constexpr int An = 2, Bn = 8, Tn = 2048, Cn = 1024, Dn = 1024, Fn = 4096;
static constexpr int Mn = Bn * Tn; // 16384 token rows

// wave tile: MT x NT WMMA tiles (16x16 each) -> 64 x 32 output per wave
static constexpr int MT = 4, NT = 2;
// block: 256 thr = 8 waves arranged 2 (M) x 4 (N) -> 128 x 128 block tile
static constexpr int BLK_M = 128, BLK_N = 128;

union Frag { v16bf v; u32x4 u[2]; };

// A fragment: 16x32 bf16.  lane L: M = L&15; half=L>>4; elems 0..7 -> K=half*8+e,
// elems 8..15 -> K=16+half*8+(e-8).  Two contiguous 8-elem runs -> 2x b128.
__device__ __forceinline__ void load_afrag(Frag& f, const bf16* __restrict__ row, int ka) {
  f.u[0] = *(const u32x4*)(row + ka);
  f.u[1] = *(const u32x4*)(row + ka + 16);
}
// B fragment: 32x16 bf16. lane L: N = L&15; half=L>>4; elems 0..7 -> K=half*8+e,
// elems 8..15 -> K=16+half*8+(e-8).  Two contiguous 8-elem runs -> 2x b128.
__device__ __forceinline__ void load_bfrag(Frag& f, const bf16* __restrict__ row, int kb) {
  f.u[0] = *(const u32x4*)(row + kb);
  f.u[1] = *(const u32x4*)(row + kb + 16);
}

__device__ __forceinline__ v8f wmma_bf16(v16bf a, v16bf b, v8f c) {
  return __builtin_amdgcn_wmma_f32_16x16x32_bf16(false, a, false, b, (short)0, c, false, false);
}

// Core K-loop: acc += A[m0.., :] * B[n0.., :]^T   (both row-major, K contiguous)
__device__ __forceinline__ void wmma_core(const bf16* __restrict__ A,
                                          const bf16* __restrict__ Bw,
                                          int K, int m0, int n0, int l16, int half,
                                          v8f (&acc)[MT][NT]) {
  const bf16* ar[MT];
  const bf16* br[NT];
#pragma unroll
  for (int i = 0; i < MT; ++i) ar[i] = A + (size_t)(m0 + i * 16 + l16) * K;
#pragma unroll
  for (int j = 0; j < NT; ++j) br[j] = Bw + (size_t)(n0 + j * 16 + l16) * K;
  for (int k0 = 0; k0 < K; k0 += 32) {
    Frag af[MT], bfr[NT];
#pragma unroll
    for (int i = 0; i < MT; ++i) load_afrag(af[i], ar[i], k0 + half * 8);
#pragma unroll
    for (int j = 0; j < NT; ++j) load_bfrag(bfr[j], br[j], k0 + half * 8);
#pragma unroll
    for (int i = 0; i < MT; ++i)
#pragma unroll
      for (int j = 0; j < NT; ++j)
        acc[i][j] = wmma_bf16(af[i].v, bfr[j].v, acc[i][j]);
  }
}

// ---------------------------------------------------------------------------
// elementwise / LN kernels
// ---------------------------------------------------------------------------

__global__ __launch_bounds__(256) void k_cvt_bf16(const float* __restrict__ s,
                                                  bf16* __restrict__ d, int n4) {
  int i = blockIdx.x * 256 + threadIdx.x;
  if (i >= n4) return;
  f32x4 v = *(const f32x4*)(s + (size_t)i * 4);
  bf16x4 o;
  o.x = (bf16)v.x; o.y = (bf16)v.y; o.z = (bf16)v.z; o.w = (bf16)v.w;
  *(bf16x4*)(d + (size_t)i * 4) = o;
}

// One 256-thread block per token row (C=1024 -> one float4 per thread).
__global__ __launch_bounds__(256) void k_layernorm(
    const float* __restrict__ x, const float* __restrict__ w, const float* __restrict__ b,
    float* __restrict__ y, float* __restrict__ xxx /*nullable*/, float* __restrict__ lastOut /*nullable*/) {
  int row = blockIdx.x;                // b*Tn + t
  int bi = row / Tn, t = row % Tn;
  int c = threadIdx.x * 4;
  f32x4 v = *(const f32x4*)(x + (size_t)row * Cn + c);
  float s  = v.x + v.y + v.z + v.w;
  float ss = v.x * v.x + v.y * v.y + v.z * v.z + v.w * v.w;
  __shared__ float red[512];
  red[threadIdx.x] = s; red[256 + threadIdx.x] = ss;
  __syncthreads();
  for (int o = 128; o > 0; o >>= 1) {
    if ((int)threadIdx.x < o) {
      red[threadIdx.x]       += red[threadIdx.x + o];
      red[256 + threadIdx.x] += red[256 + threadIdx.x + o];
    }
    __syncthreads();
  }
  float mean = red[0]   * (1.0f / Cn);
  float var  = red[256] * (1.0f / Cn) - mean * mean;
  float rstd = rsqrtf(var + 1e-5f);
  f32x4 wv = *(const f32x4*)(w + c);
  f32x4 bv = *(const f32x4*)(b + c);
  f32x4 o;
  o.x = (v.x - mean) * rstd * wv.x + bv.x;
  o.y = (v.y - mean) * rstd * wv.y + bv.y;
  o.z = (v.z - mean) * rstd * wv.z + bv.z;
  o.w = (v.w - mean) * rstd * wv.w + bv.w;
  *(f32x4*)(y + (size_t)row * Cn + c) = o;
  if (xxx) {  // xxx[a, b, t+2, :] = x_ln1 for a=0,1
    *(f32x4*)(xxx + ((size_t)(0 * Bn + bi) * (Tn + 2) + (t + 2)) * Cn + c) = o;
    *(f32x4*)(xxx + ((size_t)(1 * Bn + bi) * (Tn + 2) + (t + 2)) * Cn + c) = o;
  }
  if (lastOut && t == Tn - 1) *(f32x4*)(lastOut + (size_t)bi * Cn + c) = o;
}

// xxx[a,b,0:2,:] = att_shift_state   (A*B*2*C = 32768 elements)
__global__ __launch_bounds__(256) void k_copy_shift(const float* __restrict__ s,
                                                    float* __restrict__ xxx) {
  int i = blockIdx.x * 256 + threadIdx.x;
  if (i >= An * Bn * 2 * Cn) return;
  int c = i % Cn; int r = i / Cn;    // r = (a*Bn+b)*2 + srow
  int srow = r % 2; int ab = r / 2;
  xxx[((size_t)ab * (Tn + 2) + srow) * Cn + c] = s[i];
}

// token-shift (by 2) + per-head mix -> xk/xv/xr bf16, layout (A, B*T, C)
__global__ __launch_bounds__(256) void k_att_blend(
    const float* __restrict__ xln, const float* __restrict__ shift,
    const float* __restrict__ mk, const float* __restrict__ mv, const float* __restrict__ mr,
    bf16* __restrict__ xk, bf16* __restrict__ xv, bf16* __restrict__ xr) {
  size_t i = (size_t)blockIdx.x * 256 + threadIdx.x;   // over A*B*T*(C/4)
  int c4 = (int)(i % (Cn / 4)); size_t r = i / (Cn / 4);
  int t = (int)(r % Tn); r /= Tn;
  int b = (int)(r % Bn); int a = (int)(r / Bn);
  int c = c4 * 4;
  f32x4 xc = *(const f32x4*)(xln + ((size_t)(b * Tn + t)) * Cn + c);
  f32x4 xp = (t >= 2)
      ? *(const f32x4*)(xln + ((size_t)(b * Tn + t - 2)) * Cn + c)
      : *(const f32x4*)(shift + ((size_t)((a * Bn + b) * 2 + t)) * Cn + c);
  f32x4 k4 = *(const f32x4*)(mk + (size_t)a * Cn + c);
  f32x4 v4 = *(const f32x4*)(mv + (size_t)a * Cn + c);
  f32x4 r4 = *(const f32x4*)(mr + (size_t)a * Cn + c);
  bf16x4 ok, ov, orr;
  ok.x  = (bf16)(xc.x * k4.x + xp.x * (1.f - k4.x));
  ok.y  = (bf16)(xc.y * k4.y + xp.y * (1.f - k4.y));
  ok.z  = (bf16)(xc.z * k4.z + xp.z * (1.f - k4.z));
  ok.w  = (bf16)(xc.w * k4.w + xp.w * (1.f - k4.w));
  ov.x  = (bf16)(xc.x * v4.x + xp.x * (1.f - v4.x));
  ov.y  = (bf16)(xc.y * v4.y + xp.y * (1.f - v4.y));
  ov.z  = (bf16)(xc.z * v4.z + xp.z * (1.f - v4.z));
  ov.w  = (bf16)(xc.w * v4.w + xp.w * (1.f - v4.w));
  orr.x = (bf16)(xc.x * r4.x + xp.x * (1.f - r4.x));
  orr.y = (bf16)(xc.y * r4.y + xp.y * (1.f - r4.y));
  orr.z = (bf16)(xc.z * r4.z + xp.z * (1.f - r4.z));
  orr.w = (bf16)(xc.w * r4.w + xp.w * (1.f - r4.w));
  *(bf16x4*)(xk + i * 4) = ok;
  *(bf16x4*)(xv + i * 4) = ov;
  *(bf16x4*)(xr + i * 4) = orr;
}

// FFN token-shift (by 1) + mix -> fxk/fxr bf16 (B*T, C)
__global__ __launch_bounds__(256) void k_ffn_blend(
    const float* __restrict__ xln, const float* __restrict__ fshift,
    const float* __restrict__ mk, const float* __restrict__ mr,
    bf16* __restrict__ fxk, bf16* __restrict__ fxr) {
  size_t i = (size_t)blockIdx.x * 256 + threadIdx.x;   // over B*T*(C/4)
  int c4 = (int)(i % (Cn / 4)); size_t r = i / (Cn / 4);
  int t = (int)(r % Tn); int b = (int)(r / Tn);
  int c = c4 * 4;
  f32x4 xc = *(const f32x4*)(xln + ((size_t)(b * Tn + t)) * Cn + c);
  f32x4 xp = (t >= 1)
      ? *(const f32x4*)(xln + ((size_t)(b * Tn + t - 1)) * Cn + c)
      : *(const f32x4*)(fshift + (size_t)b * Cn + c);
  f32x4 k4 = *(const f32x4*)(mk + c);
  f32x4 r4 = *(const f32x4*)(mr + c);
  bf16x4 ok, orr;
  ok.x  = (bf16)(xc.x * k4.x + xp.x * (1.f - k4.x));
  ok.y  = (bf16)(xc.y * k4.y + xp.y * (1.f - k4.y));
  ok.z  = (bf16)(xc.z * k4.z + xp.z * (1.f - k4.z));
  ok.w  = (bf16)(xc.w * k4.w + xp.w * (1.f - k4.w));
  orr.x = (bf16)(xc.x * r4.x + xp.x * (1.f - r4.x));
  orr.y = (bf16)(xc.y * r4.y + xp.y * (1.f - r4.y));
  orr.z = (bf16)(xc.z * r4.z + xp.z * (1.f - r4.z));
  orr.w = (bf16)(xc.w * r4.w + xp.w * (1.f - r4.w));
  *(bf16x4*)(fxk + i * 4) = ok;
  *(bf16x4*)(fxr + i * 4) = orr;
}

// Sequential WKV recurrence, one thread per (a,b,d) channel; y overwrites kbuf.
__global__ __launch_bounds__(256) void k_wkv_scan(
    const float* __restrict__ kbuf, const float* __restrict__ vbuf,
    const float* __restrict__ td, const float* __restrict__ tf,
    const float* __restrict__ st_in, float* __restrict__ ybuf, float* __restrict__ st_out) {
  int tid = blockIdx.x * 256 + threadIdx.x;            // A*B*D
  if (tid >= An * Bn * Dn) return;
  int d = tid % Dn; int ab = tid / Dn; int a = ab / Bn;
  float w = -__expf(td[a * Dn + d]);
  float u = tf[a * Dn + d];
  size_t base = (size_t)ab * Tn * Dn + d;
  size_t sidx = (size_t)tid * 3;
  float aa = st_in[sidx], bb = st_in[sidx + 1], pp = st_in[sidx + 2];
  for (int t = 0; t < Tn; ++t) {
    size_t off = base + (size_t)t * Dn;
    float kt = kbuf[off], vt = vbuf[off];
    float q  = fmaxf(pp, u + kt);
    float e1 = __expf(pp - q), e2 = __expf(u + kt - q);
    float y  = (e1 * aa + e2 * vt) / (e1 * bb + e2);
    ybuf[off] = y;
    float q2  = fmaxf(pp + w, kt);
    float e1b = __expf(pp + w - q2), e2b = __expf(kt - q2);
    aa = e1b * aa + e2b * vt;
    bb = e1b * bb + e2b;
    pp = q2;
  }
  st_out[sidx] = aa; st_out[sidx + 1] = bb; st_out[sidx + 2] = pp;
}

// p = bf16(sr * y), in place over sr
__global__ __launch_bounds__(256) void k_mul_p(bf16* __restrict__ sr,
                                               const float* __restrict__ y) {
  size_t i = (size_t)blockIdx.x * 256 + threadIdx.x;   // over A*M*D/4
  bf16x4 s = *(bf16x4*)(sr + i * 4);
  f32x4 yv = *(const f32x4*)(y + i * 4);
  bf16x4 o;
  o.x = (bf16)((float)s.x * yv.x);
  o.y = (bf16)((float)s.y * yv.y);
  o.z = (bf16)((float)s.z * yv.z);
  o.w = (bf16)((float)s.w * yv.w);
  *(bf16x4*)(sr + i * 4) = o;
}

// ---------------------------------------------------------------------------
// WMMA GEMMs:  out[m,n] = sum_k A[m,k]*B[n,k]   (both row-major, K contiguous)
// 8 waves (2 along M x 4 along N); wave tile 64x32 (4x2 WMMA), block 128x128
// EPI: 0=store f32, 1=sigmoid->bf16, 2=relu^2->bf16, 3=sigmoid->f32
// ---------------------------------------------------------------------------
template <int EPI>
__global__ __launch_bounds__(256) void k_gemm_tn(
    const bf16* __restrict__ Ag, const bf16* __restrict__ Bg, void* __restrict__ Og,
    int M, int N, int K, long long asA, long long asB, long long asO) {
  int a = blockIdx.z;
  int lane = threadIdx.x & 31, wave = threadIdx.x >> 5;
  int m0 = blockIdx.y * BLK_M + (wave & 1) * 64;
  int n0 = blockIdx.x * BLK_N + (wave >> 1) * 32;
  int half = lane >> 4, l16 = lane & 15;

  v8f acc[MT][NT] = {};
  wmma_core(Ag + (size_t)a * asA, Bg + (size_t)a * asB, K, m0, n0, l16, half, acc);

#pragma unroll
  for (int i = 0; i < MT; ++i)
#pragma unroll
    for (int j = 0; j < NT; ++j) {
      int n = n0 + j * 16 + l16;
#pragma unroll
      for (int r = 0; r < 8; ++r) {
        int m = m0 + i * 16 + half * 8 + r;
        size_t idx = (size_t)m * N + n;
        float val = acc[i][j][r];
        if constexpr (EPI == 0) {
          ((float*)Og + (size_t)a * asO)[idx] = val;
        } else if constexpr (EPI == 1) {
          ((bf16*)Og + (size_t)a * asO)[idx] = (bf16)(1.f / (1.f + __expf(-val)));
        } else if constexpr (EPI == 2) {
          float t = val > 0.f ? val : 0.f;
          ((bf16*)Og)[idx] = (bf16)(t * t);
        } else {
          ((float*)Og)[idx] = 1.f / (1.f + __expf(-val));
        }
      }
    }
}

// out_x[m,c] = x_in[m,c] + sum_a sum_d p[a,m,d] * Wo[a,c,d]
__global__ __launch_bounds__(256) void k_gemm_attout(
    const bf16* __restrict__ P, const bf16* __restrict__ Wo,
    const float* __restrict__ xin, float* __restrict__ ox) {
  const int N = Cn, K = Dn;
  int lane = threadIdx.x & 31, wave = threadIdx.x >> 5;
  int m0 = blockIdx.y * BLK_M + (wave & 1) * 64;
  int n0 = blockIdx.x * BLK_N + (wave >> 1) * 32;
  int half = lane >> 4, l16 = lane & 15;
  v8f acc[MT][NT] = {};
  for (int a = 0; a < An; ++a)
    wmma_core(P + (size_t)a * Mn * Dn, Wo + (size_t)a * Cn * Dn, K, m0, n0, l16, half, acc);
#pragma unroll
  for (int i = 0; i < MT; ++i)
#pragma unroll
    for (int j = 0; j < NT; ++j) {
      int n = n0 + j * 16 + l16;
#pragma unroll
      for (int r = 0; r < 8; ++r) {
        size_t idx = (size_t)(m0 + i * 16 + half * 8 + r) * N + n;
        ox[idx] = xin[idx] + acc[i][j][r];
      }
    }
}

// out_x[m,c] += sig_r[m,c] * (sum_f kf[m,f] * Wv[c,f])
__global__ __launch_bounds__(256) void k_gemm_ffnout(
    const bf16* __restrict__ KF, const bf16* __restrict__ Wv,
    const float* __restrict__ sigr, float* __restrict__ ox) {
  const int N = Cn, K = Fn;
  int lane = threadIdx.x & 31, wave = threadIdx.x >> 5;
  int m0 = blockIdx.y * BLK_M + (wave & 1) * 64;
  int n0 = blockIdx.x * BLK_N + (wave >> 1) * 32;
  int half = lane >> 4, l16 = lane & 15;
  v8f acc[MT][NT] = {};
  wmma_core(KF, Wv, K, m0, n0, l16, half, acc);
#pragma unroll
  for (int i = 0; i < MT; ++i)
#pragma unroll
    for (int j = 0; j < NT; ++j) {
      int n = n0 + j * 16 + l16;
#pragma unroll
      for (int r = 0; r < 8; ++r) {
        size_t idx = (size_t)(m0 + i * 16 + half * 8 + r) * N + n;
        ox[idx] += sigr[idx] * acc[i][j][r];
      }
    }
}

// ---------------------------------------------------------------------------
// host orchestration
// ---------------------------------------------------------------------------
extern "C" void kernel_launch(void* const* d_in, const int* in_sizes, int n_in,
                              void* d_out, int out_size, void* d_ws, size_t ws_size,
                              hipStream_t stream) {
  (void)in_sizes; (void)n_in; (void)out_size; (void)ws_size;
  const float* x      = (const float*)d_in[0];
  const float* ln1w   = (const float*)d_in[1];
  const float* ln1b   = (const float*)d_in[2];
  const float* ln2w   = (const float*)d_in[3];
  const float* ln2b   = (const float*)d_in[4];
  const float* tdec   = (const float*)d_in[5];
  const float* tfirst = (const float*)d_in[6];
  const float* amk    = (const float*)d_in[7];
  const float* amv    = (const float*)d_in[8];
  const float* amr    = (const float*)d_in[9];
  const float* aWk    = (const float*)d_in[10];
  const float* aWv    = (const float*)d_in[11];
  const float* aWr    = (const float*)d_in[12];
  const float* aWo    = (const float*)d_in[13];
  const float* ashift = (const float*)d_in[14];
  const float* awkv   = (const float*)d_in[15];
  const float* fmk    = (const float*)d_in[16];
  const float* fmr    = (const float*)d_in[17];
  const float* fWk    = (const float*)d_in[18];
  const float* fWr    = (const float*)d_in[19];
  const float* fWv    = (const float*)d_in[20];
  const float* fshift = (const float*)d_in[21];

  float* out_x    = (float*)d_out;                                // (B,T,C)
  float* out_xxx  = out_x + (size_t)Mn * Cn;                      // (A,B,T+2,C)
  float* out_wkv  = out_xxx + (size_t)An * Bn * (Tn + 2) * Cn;    // (A,B,D,3)
  float* out_last = out_wkv + (size_t)An * Bn * Dn * 3;           // (B,C)

  // workspace layout (regions reused across phases)
  char* ws = (char*)d_ws;
  const size_t S1 = (size_t)Mn * Cn * 4;        // 64 MiB  x_ln1 / x_ln2
  const size_t S2 = (size_t)An * Mn * Cn * 2;   // 64 MiB  per blended bf16 tensor
  const size_t S4 = (size_t)An * Mn * Dn * 4;   // 128 MiB k / v (f32)
  const size_t S6 = (size_t)An * Mn * Dn * 2;   // 64 MiB  sr bf16 (== M*C*4 for sigr)
  float* xln  = (float*)(ws);                    // later x_ln2
  bf16*  xk   = (bf16*)(ws + S1);
  bf16*  xv   = (bf16*)(ws + S1 + S2);
  bf16*  xr   = (bf16*)(ws + S1 + 2 * S2);
  float* kbuf = (float*)(ws + S1 + 3 * S2);      // later y (in place)
  float* vbuf = (float*)(ws + S1 + 3 * S2 + S4);
  bf16*  srb  = (bf16*)(ws + S1 + 3 * S2 + 2 * S4);
  char*  wsp  = ws + S1 + 3 * S2 + 2 * S4 + S6;
  bf16* wkb  = (bf16*)(wsp);                     wsp += (size_t)An * Dn * Cn * 2;
  bf16* wvb  = (bf16*)(wsp);                     wsp += (size_t)An * Dn * Cn * 2;
  bf16* wrb  = (bf16*)(wsp);                     wsp += (size_t)An * Dn * Cn * 2;
  bf16* wob  = (bf16*)(wsp);                     wsp += (size_t)An * Cn * Dn * 2;
  bf16* fwkb = (bf16*)(wsp);                     wsp += (size_t)Fn * Cn * 2;
  bf16* fwrb = (bf16*)(wsp);                     wsp += (size_t)Cn * Cn * 2;
  bf16* fwvb = (bf16*)(wsp);
  // phase-2 aliases
  bf16*  kfb  = (bf16*)xk;                       // (M,F) bf16 spans xk+xv regions
  bf16*  fxk  = (bf16*)xr;                       // (M,C) bf16
  bf16*  fxr  = fxk + (size_t)Mn * Cn;           // (M,C) bf16
  float* sigr = (float*)srb;                     // (M,C) f32

  auto cvt = [&](const float* s, bf16* d, size_t n) {
    int n4 = (int)(n / 4);
    k_cvt_bf16<<<(n4 + 255) / 256, 256, 0, stream>>>(s, d, n4);
  };
  cvt(aWk, wkb, (size_t)An * Dn * Cn);
  cvt(aWv, wvb, (size_t)An * Dn * Cn);
  cvt(aWr, wrb, (size_t)An * Dn * Cn);
  cvt(aWo, wob, (size_t)An * Cn * Dn);
  cvt(fWk, fwkb, (size_t)Fn * Cn);
  cvt(fWr, fwrb, (size_t)Cn * Cn);
  cvt(fWv, fwvb, (size_t)Cn * Fn);

  // attention path
  k_layernorm<<<Mn, 256, 0, stream>>>(x, ln1w, ln1b, xln, out_xxx, nullptr);
  k_copy_shift<<<(An * Bn * 2 * Cn + 255) / 256, 256, 0, stream>>>(ashift, out_xxx);
  k_att_blend<<<(unsigned)((size_t)An * Mn * (Cn / 4) / 256), 256, 0, stream>>>(
      xln, ashift, amk, amv, amr, xk, xv, xr);

  dim3 gKV(Dn / BLK_N, Mn / BLK_M, An);          // (8, 128, 2)
  const long long sA = (long long)Mn * Cn, sB = (long long)Dn * Cn, sO = (long long)Mn * Dn;
  k_gemm_tn<0><<<gKV, 256, 0, stream>>>(xk, wkb, (void*)kbuf, Mn, Dn, Cn, sA, sB, sO);
  k_gemm_tn<0><<<gKV, 256, 0, stream>>>(xv, wvb, (void*)vbuf, Mn, Dn, Cn, sA, sB, sO);
  k_gemm_tn<1><<<gKV, 256, 0, stream>>>(xr, wrb, (void*)srb, Mn, Dn, Cn, sA, sB, sO);

  k_wkv_scan<<<(An * Bn * Dn) / 256, 256, 0, stream>>>(kbuf, vbuf, tdec, tfirst, awkv,
                                                       kbuf, out_wkv);
  k_mul_p<<<(unsigned)((size_t)An * Mn * Dn / 4 / 256), 256, 0, stream>>>(srb, kbuf);
  k_gemm_attout<<<dim3(Cn / BLK_N, Mn / BLK_M, 1), 256, 0, stream>>>(srb, wob, x, out_x);

  // FFN path
  k_layernorm<<<Mn, 256, 0, stream>>>(out_x, ln2w, ln2b, xln, nullptr, out_last);
  k_ffn_blend<<<(unsigned)((size_t)Mn * (Cn / 4) / 256), 256, 0, stream>>>(
      xln, fshift, fmk, fmr, fxk, fxr);
  k_gemm_tn<2><<<dim3(Fn / BLK_N, Mn / BLK_M, 1), 256, 0, stream>>>(
      fxk, fwkb, (void*)kfb, Mn, Fn, Cn, 0, 0, 0);
  k_gemm_tn<3><<<dim3(Cn / BLK_N, Mn / BLK_M, 1), 256, 0, stream>>>(
      fxr, fwrb, (void*)sigr, Mn, Cn, Cn, 0, 0, 0);
  k_gemm_ffnout<<<dim3(Cn / BLK_N, Mn / BLK_M, 1), 256, 0, stream>>>(kfb, fwvb, sigr, out_x);
}
